// TritonFusedMBConvBlock_36146444763152
// MI455X (gfx1250) — matrix operand order repaired and checked by reading, not compiled, not run on hardware
//
#include <hip/hip_runtime.h>
#include <hip/hip_bf16.h>

typedef __attribute__((ext_vector_type(16))) __bf16 v16bf;
typedef __attribute__((ext_vector_type(8)))  float  v8f;
typedef __attribute__((ext_vector_type(4)))  unsigned int u32x4;
typedef __attribute__((ext_vector_type(8)))  int  i32x8;
typedef __attribute__((ext_vector_type(4)))  int  i32x4;

#define IN_C   64
#define EXP_C  256
#define SE_CH  64
#define OUT_C  64
#define HH     112
#define WW     112
#define HWPIX  (HH*WW)          // 12544
#define BATCH  32

// workspace layout (bytes)
#define H_WS_BYTES ((size_t)BATCH*HWPIX*EXP_C*2)      // h in bf16, pixel-major [b][pix][e]
#define POOL_OFF   H_WS_BYTES
#define SE_OFF     (POOL_OFF + (size_t)BATCH*EXP_C*4)

// K1 LDS map (bytes)
#define X_OFF      0          // 336 px * 64 ch bf16              = 43008
#define W1_OFF     43008      // 16mt*2ks*32lane*16 bf16          = 32768
#define HST_OFF    75776      // 8 waves * 16ch * 324px bf16      = 82944
#define OUTL_OFF   158720     // 256 px * 256 ch bf16             = 131072
#define B1_OFF     289792     // 256 f32                          = 1024
#define POOLL_OFF  290816     // 256 f32                          = 1024
#define K1_LDS     291840

// K3 LDS map (bytes)
#define K3_W5_OFF  0          // 4mt*8ks*32lane*16 bf16           = 32768
#define K3_H_OFF   32768      // 128 px * 256 ch bf16             = 65536
#define K3_OUT_OFF 98304      // 64 o * 128 px f32                = 32768
#define K3_LDS     131072

// ---- TDM descriptor helpers (ISA cdna5 ch.8): 2D tile, data_size=8B ----
// group0: [1:0]=count=1, [63:32]=lds_addr, [120:64]=global_addr, [127:126]=type=2
__device__ __forceinline__ u32x4 tdm_g0(unsigned lds_addr, unsigned long long gaddr) {
    u32x4 g0;
    g0[0] = 1u;                                   // count=1, user mode
    g0[1] = lds_addr;
    g0[2] = (unsigned)gaddr;
    g0[3] = ((unsigned)(gaddr >> 32) & 0x1FFFFFFu) | (2u << 30);   // type=2 ("image")
    return g0;
}
// group1: data_size=8B(3); tensor_dim0/1, tile_dim0/1, tensor_dim0_stride (8B units)
__device__ __forceinline__ i32x8 tdm_g1(unsigned tensor_d0, unsigned tensor_d1,
                                        unsigned tile_d0, unsigned tile_d1,
                                        unsigned stride0) {
    i32x8 g1;
    g1[0] = 3 << 16;                              // workgroup_mask=0, data_size=8B
    g1[1] = (int)((tensor_d0 & 0xFFFFu) << 16);   // [63:48] = tensor_dim0 lo16
    g1[2] = (int)((tensor_d0 >> 16) | ((tensor_d1 & 0xFFFFu) << 16));
    g1[3] = (int)((tensor_d1 >> 16) | (tile_d0 << 16));   // [127:112] tile_dim0
    g1[4] = (int)tile_d1;                         // [143:128] tile_dim1, tile_dim2=0
    g1[5] = (int)stride0;                         // tensor_dim0_stride lo32
    g1[6] = 0;
    g1[7] = 0;
    return g1;
}

// ---------------- kernel 0: zero the pool accumulator ----------------
__global__ void k0_zero(float* __restrict__ pool_g) {
    pool_g[blockIdx.x * 256 + threadIdx.x] = 0.f;
}

// ---------------- kernel 1: expand 1x1 (WMMA bf16) + ReLU + DW 3x3 + ReLU + pool partials
// grid (7,7,32), block 256, dynamic LDS K1_LDS
__global__ __launch_bounds__(256)
void k1_expand_dw(const float* __restrict__ x,
                  const float* __restrict__ w1,
                  const float* __restrict__ w2,
                  const float* __restrict__ b1,
                  const float* __restrict__ b2,
                  __bf16* __restrict__ h_ws,
                  float* __restrict__ pool_g)
{
    extern __shared__ char smem[];
    __bf16* x_lds    = (__bf16*)(smem + X_OFF);
    __bf16* w1_lds   = (__bf16*)(smem + W1_OFF);
    __bf16* hstage   = (__bf16*)(smem + HST_OFF);
    __bf16* out_lds  = (__bf16*)(smem + OUTL_OFF);
    float*  b1_lds   = (float*)(smem + B1_OFF);
    float*  pool_lds = (float*)(smem + POOLL_OFF);

    const int tid = threadIdx.x;
    const int lane = tid & 31, wave = tid >> 5;
    const int lanelo = lane & 15, lanehi = lane >> 4;

    const int tx0 = blockIdx.x * 16, ty0 = blockIdx.y * 16, b = blockIdx.z;

    if (tid < EXP_C) { pool_lds[tid] = 0.f; b1_lds[tid] = b1[tid]; }

    // stage x halo tile (18x18) as bf16, pixel-major [pix][ch]
    const float* xb = x + (size_t)b * IN_C * HWPIX;
    for (int t = tid; t < 324 * 64; t += 256) {
        int c = t / 324, p = t - c * 324;
        int hy = p / 18, hx = p - hy * 18;
        int gy = ty0 - 1 + hy, gx = tx0 - 1 + hx;
        float v = 0.f;
        if (gy >= 0 && gy < HH && gx >= 0 && gx < WW)
            v = xb[(size_t)c * HWPIX + gy * WW + gx];
        x_lds[p * 64 + c] = (__bf16)v;
    }
    for (int t = tid; t < 12 * 64; t += 256)                   // zero pad pixels 324..335
        x_lds[(324 + (t >> 6)) * 64 + (t & 63)] = (__bf16)0.f;

    // pre-swizzle w1 (256x64 fp32) into A-fragment layout (contiguous k: lane half h holds k 16h..16h+15)
    for (int s = tid; s < 16 * 2 * 32; s += 256) {
        int l = s & 31, ks = (s >> 5) & 1, mt = s >> 6;
        int m = mt * 16 + (l & 15);
        int kbase = ks * 32 + ((l >> 4) ? 16 : 0);
        #pragma unroll
        for (int e = 0; e < 16; ++e) {
            int k = kbase + e;
            w1_lds[s * 16 + e] = (__bf16)w1[m * IN_C + k];
        }
    }
    __syncthreads();

    __bf16* hst = hstage + wave * 16 * 324;                    // per-wave staging
    for (int mi = 0; mi < 2; ++mi) {
        const int mt = wave + mi * 8;                          // 16 exp-channel tile
        v16bf a0 = *(const v16bf*)&w1_lds[((mt * 2 + 0) * 32 + lane) * 16];
        v16bf a1 = *(const v16bf*)&w1_lds[((mt * 2 + 1) * 32 + lane) * 16];

        // software-pipelined over 21 pixel tiles (336 halo pixels)
        v16bf bb0 = *(const v16bf*)&x_lds[lanelo * 64 + lanehi * 16];
        v16bf bb1 = *(const v16bf*)&x_lds[lanelo * 64 + 32 + lanehi * 16];
        for (int nt = 0; nt < 21; ++nt) {
            const int pnx = ((nt < 20) ? nt + 1 : nt) * 16 + lanelo;
            v16bf nb0 = *(const v16bf*)&x_lds[pnx * 64 + lanehi * 16];
            v16bf nb1 = *(const v16bf*)&x_lds[pnx * 64 + 32 + lanehi * 16];

            v8f c = {};
            c = __builtin_amdgcn_wmma_f32_16x16x32_bf16(false, a0, false, bb0, (short)0, c, false, false);
            c = __builtin_amdgcn_wmma_f32_16x16x32_bf16(false, a1, false, bb1, (short)0, c, false, false);

            const int pix = nt * 16 + lanelo;
            if (pix < 324) {
                int hy = pix / 18, hx = pix - hy * 18;
                int gy = ty0 - 1 + hy, gx = tx0 - 1 + hx;
                bool valid = (gy >= 0 && gy < HH && gx >= 0 && gx < WW);
                #pragma unroll
                for (int r = 0; r < 8; ++r) {
                    int ml = r + lanehi * 8;
                    float v = c[r] + b1_lds[mt * 16 + ml];
                    v = valid ? (v > 0.f ? v : 0.f) : 0.f;     // DW conv zero-pads h
                    hst[ml * 324 + pix] = (__bf16)v;
                }
            }
            bb0 = nb0; bb1 = nb1;
        }

        // depthwise 3x3 on the staged 16ch x 18x18 tile (VALU + LDS)
        for (int i = 0; i < 8; ++i) {
            int tt = lane + 32 * i;                            // 16 ch * 16 rows tasks
            int ch = tt >> 4, oy = tt & 15;
            int e = mt * 16 + ch;
            float wk[9];
            #pragma unroll
            for (int j = 0; j < 9; ++j) wk[j] = w2[e * 9 + j];
            float bias = b2[e];
            const __bf16* hrow = &hst[ch * 324];
            float rowsum = 0.f;
            #pragma unroll
            for (int ox = 0; ox < 16; ++ox) {
                float acc = bias;
                #pragma unroll
                for (int ky = 0; ky < 3; ++ky)
                    #pragma unroll
                    for (int kx = 0; kx < 3; ++kx)
                        acc += wk[ky * 3 + kx] * (float)hrow[(oy + ky) * 18 + (ox + kx)];
                float v = acc > 0.f ? acc : 0.f;
                rowsum += v;
                out_lds[(oy * 16 + ox) * EXP_C + e] = (__bf16)v;   // stage for TDM
            }
            atomicAdd(&pool_lds[e], rowsum);                   // ds_add_f32
        }
    }
    __syncthreads();

    // TDM writeback: one descriptor, 2D tile = 16 rows x 8KB, row stride 57344B
    if (tid < 32) {                                            // wave 0 only (TDM ignores EXEC)
        unsigned long long ga = (unsigned long long)((const char*)h_ws
                              + ((size_t)b * HWPIX + (size_t)ty0 * WW + tx0) * EXP_C * 2);
        u32x4 g0 = tdm_g0((unsigned)(size_t)out_lds, ga);
        i32x8 g1 = tdm_g1(/*tensor_d0=*/WW * EXP_C * 2 / 8, /*tensor_d1=*/16,
                          /*tile_d0=*/16 * EXP_C * 2 / 8, /*tile_d1=*/16,
                          /*stride0=*/WW * EXP_C * 2 / 8);
        i32x4 z4 = {};
        i32x8 z8 = {};
        __builtin_amdgcn_tensor_store_from_lds(g0, g1, z4, z4, z8, 0);
        __builtin_amdgcn_s_wait_tensorcnt(0);                  // LDS freed at workgroup end
    }
    if (tid < EXP_C) atomicAdd(&pool_g[b * EXP_C + tid], pool_lds[tid]);
}

// ---------------- kernel 2: SE (mean -> FC+ReLU -> FC+sigmoid) ----------------
__global__ __launch_bounds__(256)
void k2_se(const float* __restrict__ w3, const float* __restrict__ w4,
           const float* __restrict__ b3, const float* __restrict__ b4,
           const float* __restrict__ pool_g, float* __restrict__ se_g)
{
    __shared__ float mean_lds[EXP_C];
    __shared__ float se1_lds[SE_CH];
    const int b = blockIdx.x, tid = threadIdx.x;
    if (tid < EXP_C) mean_lds[tid] = pool_g[b * EXP_C + tid] * (1.f / (float)HWPIX);
    __syncthreads();
    if (tid < SE_CH) {
        float acc = b3[tid];
        for (int e = 0; e < EXP_C; ++e) acc += w3[tid * EXP_C + e] * mean_lds[e];
        se1_lds[tid] = acc > 0.f ? acc : 0.f;
    }
    __syncthreads();
    if (tid < EXP_C) {
        float acc = b4[tid];
        for (int s = 0; s < SE_CH; ++s) acc += w4[tid * SE_CH + s] * se1_lds[s];
        se_g[b * EXP_C + tid] = 1.f / (1.f + __expf(-acc));
    }
}

// ---------------- kernel 3: project 1x1 (WMMA bf16), SE folded into weights ----------------
// grid 32*98, block 256, dynamic LDS K3_LDS
__global__ __launch_bounds__(256)
void k3_project(const __bf16* __restrict__ h_ws,
                const float* __restrict__ w5,
                const float* __restrict__ b5,
                const float* __restrict__ se_g,
                float* __restrict__ out)
{
    extern __shared__ char smem[];
    __bf16* w5_lds  = (__bf16*)(smem + K3_W5_OFF);
    __bf16* h_lds   = (__bf16*)(smem + K3_H_OFF);
    float*  out_lds = (float*)(smem + K3_OUT_OFF);

    const int tid = threadIdx.x;
    const int lane = tid & 31, wave = tid >> 5;
    const int lanelo = lane & 15, lanehi = lane >> 4;
    const int b = blockIdx.x / 98, pt = blockIdx.x % 98;
    const int p0 = pt * 128;

    // TDM fill: one descriptor, contiguous 64KB h tile -> LDS (tracked by TENSORcnt)
    if (tid < 32) {                                            // wave 0 only
        unsigned long long ga = (unsigned long long)((const char*)h_ws
                              + ((size_t)b * HWPIX + p0) * EXP_C * 2);
        u32x4 g0 = tdm_g0((unsigned)(size_t)h_lds, ga);
        i32x8 g1 = tdm_g1(/*tensor_d0=*/65536 / 8, /*tensor_d1=*/1,
                          /*tile_d0=*/65536 / 8, /*tile_d1=*/1,
                          /*stride0=*/65536 / 8);
        i32x4 z4 = {};
        i32x8 z8 = {};
        __builtin_amdgcn_tensor_load_to_lds(g0, g1, z4, z4, z8, 0);
    }

    // scale w5 columns by SE and pre-swizzle into A-fragment layout (overlaps TDM)
    const float* seb = se_g + b * EXP_C;
    for (int s = tid; s < 4 * 8 * 32; s += 256) {
        int l = s & 31, ks = (s >> 5) & 7, mt = s >> 8;
        int m = mt * 16 + (l & 15);
        int kbase = ks * 32 + ((l >> 4) ? 16 : 0);
        #pragma unroll
        for (int e = 0; e < 16; ++e) {
            int k = kbase + e;
            w5_lds[s * 16 + e] = (__bf16)(w5[m * EXP_C + k] * seb[k]);
        }
    }
    if (tid < 32) __builtin_amdgcn_s_wait_tensorcnt(0);
    __syncthreads();

    // 4x8 output tiles of 16x16, 8 waves -> 4 tiles each; K=256 -> 8 wmma per tile
    for (int i = 0; i < 4; ++i) {
        int tile = wave * 4 + i;
        int mt = tile >> 3, nt = tile & 7;
        v8f c = {};
        #pragma unroll
        for (int ks = 0; ks < 8; ++ks) {
            v16bf a  = *(const v16bf*)&w5_lds[((mt * 8 + ks) * 32 + lane) * 16];
            v16bf bb = *(const v16bf*)&h_lds[(nt * 16 + lanelo) * EXP_C + ks * 32 + lanehi * 16];
            c = __builtin_amdgcn_wmma_f32_16x16x32_bf16(false, a, false, bb, (short)0, c, false, false);
        }
        int px = nt * 16 + lanelo;
        #pragma unroll
        for (int r = 0; r < 8; ++r) {
            int o = mt * 16 + r + lanehi * 8;
            out_lds[o * 128 + px] = c[r] + b5[o];
        }
    }
    __syncthreads();

    // coalesced async writeback: 64 output-channel rows x 512B contiguous chunks
    {
        unsigned lds0 = (unsigned)(size_t)out_lds;
        const char* gb = (const char*)out + ((size_t)b * OUT_C * HWPIX + p0) * 4;
        #pragma unroll
        for (int i = 0; i < 8; ++i) {
            int j = tid + i * 256;                 // 2048 x 16B transfers
            int o = j >> 5;                        // 32 chunks per o-row
            int chunk = j & 31;
            unsigned la = lds0 + (unsigned)j * 16;
            unsigned long long ga = (unsigned long long)(gb
                                   + (size_t)o * (HWPIX * 4) + (size_t)chunk * 16);
            asm volatile("global_store_async_from_lds_b128 %0, %1, off"
                         :: "v"(ga), "v"(la) : "memory");
        }
        asm volatile("s_wait_asynccnt 0" ::: "memory");
    }
}

extern "C" void kernel_launch(void* const* d_in, const int* in_sizes, int n_in,
                              void* d_out, int out_size, void* d_ws, size_t ws_size,
                              hipStream_t stream) {
    const float* x  = (const float*)d_in[0];
    const float* w1 = (const float*)d_in[1];
    const float* w2 = (const float*)d_in[2];
    const float* w3 = (const float*)d_in[3];
    const float* w4 = (const float*)d_in[4];
    const float* w5 = (const float*)d_in[5];
    const float* b1 = (const float*)d_in[6];
    const float* b2 = (const float*)d_in[7];
    const float* b3 = (const float*)d_in[8];
    const float* b4 = (const float*)d_in[9];
    const float* b5 = (const float*)d_in[10];
    float* out = (float*)d_out;

    __bf16* h_ws  = (__bf16*)d_ws;
    float* pool_g = (float*)((char*)d_ws + POOL_OFF);
    float* se_g   = (float*)((char*)d_ws + SE_OFF);

    k0_zero<<<dim3(BATCH), dim3(256), 0, stream>>>(pool_g);
    k1_expand_dw<<<dim3(7, 7, BATCH), dim3(256), K1_LDS, stream>>>(x, w1, w2, b1, b2, h_ws, pool_g);
    k2_se<<<dim3(BATCH), dim3(256), 0, stream>>>(w3, w4, b3, b4, pool_g, se_g);
    k3_project<<<dim3(BATCH * 98), dim3(256), K3_LDS, stream>>>(h_ws, w5, b5, se_g, out);
}
